// DeepRNN_89936615179057
// MI455X (gfx1250) — matrix operand order repaired and checked by reading, not compiled, not run on hardware
//
#include <hip/hip_runtime.h>
#include <hip/hip_bf16.h>
#include <math.h>

// ---------------------------------------------------------------------------
// Types for CDNA5 WMMA
// ---------------------------------------------------------------------------
typedef __attribute__((ext_vector_type(16))) __bf16    v16bf;
typedef __attribute__((ext_vector_type(8)))  float     v8f;
typedef __attribute__((ext_vector_type(8)))  unsigned  v8u;

#define SEQ    512
#define BATCH  64
#define DH     1024
#define MTOT   (SEQ * BATCH)          // 32768 rows for the projection GEMM
#define DH2    (DH / 2)               // u32 (bf16 pair) row pitch

// fp32 -> packed bf16x2 with round-to-nearest-even
__device__ __forceinline__ unsigned pack_bf16(float a, float b) {
    union { float f; unsigned u; } ua, ub;
    ua.f = a; ub.f = b;
    unsigned ra = ua.u + 0x7FFFu + ((ua.u >> 16) & 1u);
    unsigned rb = ub.u + 0x7FFFu + ((ub.u >> 16) & 1u);
    return (ra >> 16) | (rb & 0xFFFF0000u);
}

__device__ __forceinline__ v16bf as_bf16x16(v8u u) {
    return __builtin_bit_cast(v16bf, u);
}

__device__ __forceinline__ v8u mk8(uint4 a, uint4 b) {
    v8u r = {a.x, a.y, a.z, a.w, b.x, b.y, b.z, b.w};
    return r;
}

__device__ __forceinline__ v8f wmma_bf16(v8u a, v8u b, v8f c) {
    return __builtin_amdgcn_wmma_f32_16x16x32_bf16(
        false, as_bf16x16(a), false, as_bf16x16(b), (short)0, c, false, false);
}

// ---------------------------------------------------------------------------
// fp32 -> bf16 conversion (pairs), grid-stride
// ---------------------------------------------------------------------------
__global__ void cvt_f32_bf16(const float* __restrict__ src,
                             unsigned* __restrict__ dst, unsigned npairs) {
    unsigned i = blockIdx.x * blockDim.x + threadIdx.x;
    unsigned stride = gridDim.x * blockDim.x;
    const float2* s2 = (const float2*)src;
    for (; i < npairs; i += stride) {
        float2 f = s2[i];
        dst[i] = pack_bf16(f.x, f.y);
    }
}

// bias = bih + bhh
__global__ void bias_sum(const float* __restrict__ a, const float* __restrict__ b,
                         float* __restrict__ c) {
    int i = blockIdx.x * blockDim.x + threadIdx.x;
    if (i < DH) c[i] = a[i] + b[i];
}

// ---------------------------------------------------------------------------
// Input projection: xp[M,DH] = Xbf[M,DH] * Wbf[DH,DH]^T + bias
// One wave computes a 16x64 strip (1 M-tile x 4 N-tiles), full K=1024.
// Ping-pong software pipeline (unroll-by-2): each iteration loads one
// fragment set while consuming the other, with no register copies.
// A fragment (16x32 bf16): lane m=lane&15, hi=lane>>4;
//   u32 idx: v<4 -> kb2 + hi*4 + v ; v>=4 -> kb2 + 8 + hi*4 + (v-4)
// B fragment (32x16 bf16): lane n; u32 idx: same as A (v<4 -> kb2 + hi*4 + v ; v>=4 -> kb2 + 8 + hi*4 + (v-4))
// D (f32 16x16): M = r + 8*hi, N = lane&15
// ---------------------------------------------------------------------------
__global__ void proj_gemm(const unsigned short* __restrict__ xbf,
                          const unsigned short* __restrict__ wbf,
                          const float* __restrict__ bias,
                          float* __restrict__ xp) {
    const int wave = (blockIdx.x * blockDim.x + threadIdx.x) >> 5;
    const int lane = threadIdx.x & 31;
    const int ngrp = wave & 15;       // 16 column groups of 64
    const int mt   = wave >> 4;       // 2048 row tiles of 16
    if (mt >= MTOT / 16) return;

    const int l15 = lane & 15;
    const int hi  = lane >> 4;

    const uint4* xq = (const uint4*)((const unsigned*)xbf + (size_t)(mt * 16 + l15) * DH2);
    const uint4* wq0 = (const uint4*)((const unsigned*)wbf + (size_t)(ngrp * 64 +  0 + l15) * DH2);
    const uint4* wq1 = (const uint4*)((const unsigned*)wbf + (size_t)(ngrp * 64 + 16 + l15) * DH2);
    const uint4* wq2 = (const uint4*)((const unsigned*)wbf + (size_t)(ngrp * 64 + 32 + l15) * DH2);
    const uint4* wq3 = (const uint4*)((const unsigned*)wbf + (size_t)(ngrp * 64 + 48 + l15) * DH2);

    auto loadA = [&](int kq) { return mk8(xq[kq + hi], xq[kq + 2 + hi]); };
    auto loadB = [&](const uint4* wq, int kq) {
        return mk8(wq[kq + hi], wq[kq + 2 + hi]);
    };

    v8f acc0 = {}, acc1 = {}, acc2 = {}, acc3 = {};

    // phase 0 prefetch
    v8u a0  = loadA(0);
    v8u b00 = loadB(wq0, 0), b01 = loadB(wq1, 0);
    v8u b02 = loadB(wq2, 0), b03 = loadB(wq3, 0);

    for (int kb2 = 0; kb2 < DH2; kb2 += 32) {   // two K-steps of 32 bf16 each
        // prefetch phase 1 while phase 0 computes
        const int kq1 = (kb2 + 16) >> 2;
        v8u a1  = loadA(kq1);
        v8u b10 = loadB(wq0, kq1), b11 = loadB(wq1, kq1);
        v8u b12 = loadB(wq2, kq1), b13 = loadB(wq3, kq1);

        acc0 = wmma_bf16(a0, b00, acc0);
        acc1 = wmma_bf16(a0, b01, acc1);
        acc2 = wmma_bf16(a0, b02, acc2);
        acc3 = wmma_bf16(a0, b03, acc3);

        // prefetch next phase 0 while phase 1 computes
        if (kb2 + 32 < DH2) {
            const int kq2 = (kb2 + 32) >> 2;
            a0  = loadA(kq2);
            b00 = loadB(wq0, kq2); b01 = loadB(wq1, kq2);
            b02 = loadB(wq2, kq2); b03 = loadB(wq3, kq2);
        }

        acc0 = wmma_bf16(a1, b10, acc0);
        acc1 = wmma_bf16(a1, b11, acc1);
        acc2 = wmma_bf16(a1, b12, acc2);
        acc3 = wmma_bf16(a1, b13, acc3);
    }

#pragma unroll
    for (int j = 0; j < 4; ++j) {
        const v8f acc = (j == 0) ? acc0 : (j == 1) ? acc1 : (j == 2) ? acc2 : acc3;
        int col = ngrp * 64 + j * 16 + l15;
        float bv = bias[col];
#pragma unroll
        for (int r = 0; r < 8; ++r) {
            int row = mt * 16 + r + 8 * hi;
            xp[(size_t)row * DH + col] = acc[r] + bv;
        }
    }
}

// ---------------------------------------------------------------------------
// One recurrence step (in-place): out[t] = act(xp[t] + out[t-1] * Whh^T)
// Block = 4 waves; each block owns one 16x16 output tile; wave w reduces
// K in [w*256,(w+1)*256) via 8 bf16 WMMAs; partials reduced through LDS.
// H (fp32) is packed to bf16 in-register. Ping-pong pipelined K loop +
// prefetch of the xp tile needed by the epilogue.
// ---------------------------------------------------------------------------
__global__ void rnn_step(const unsigned short* __restrict__ whbf,
                         float* __restrict__ outbase, int t, int act) {
    __shared__ float red[4][32][8];
    const int w    = threadIdx.x >> 5;
    const int lane = threadIdx.x & 31;
    const int mt   = blockIdx.x & 3;   // 4 row tiles over batch=64
    const int nt   = blockIdx.x >> 2;  // 64 col tiles over DH
    const int l15  = lane & 15;
    const int hi   = lane >> 4;

    float* xp = outbase + (size_t)t * BATCH * DH;

    // warm the xp tile toward this WGP while the matrix work runs
    if (w == 0) {
#pragma unroll
        for (int r = 0; r < 8; ++r) {
            int row = mt * 16 + r + 8 * hi;
            __builtin_prefetch(&xp[(size_t)row * DH + nt * 16 + l15], 0, 0);
        }
    }

    v8f acc = {};
    if (t > 0) {
        const float* hrow = outbase + (size_t)(t - 1) * BATCH * DH
                                    + (size_t)(mt * 16 + l15) * DH;
        const float4* h4 = (const float4*)hrow;
        const uint4* wq = (const uint4*)((const unsigned*)whbf
                                         + (size_t)(nt * 16 + l15) * DH2);

        auto loadH = [&](int kb, float4* p) {
            p[0] = h4[(kb + hi * 8) >> 2];
            p[1] = h4[((kb + hi * 8) >> 2) + 1];
            p[2] = h4[(kb + 16 + hi * 8) >> 2];
            p[3] = h4[((kb + 16 + hi * 8) >> 2) + 1];
        };
        auto loadW = [&](int kb) {
            const int kq = kb >> 3;   // u32 idx / 4 => (kb/2)/4
            return mk8(wq[kq + hi], wq[kq + 2 + hi]);
        };
        auto packA = [&](const float4* p) {
            v8u au = {pack_bf16(p[0].x, p[0].y), pack_bf16(p[0].z, p[0].w),
                      pack_bf16(p[1].x, p[1].y), pack_bf16(p[1].z, p[1].w),
                      pack_bf16(p[2].x, p[2].y), pack_bf16(p[2].z, p[2].w),
                      pack_bf16(p[3].x, p[3].y), pack_bf16(p[3].z, p[3].w)};
            return au;
        };

        const int kbase = w * 256;
        float4 p0[4];
        loadH(kbase, p0);
        v8u w0 = loadW(kbase);

#pragma unroll
        for (int kk = 0; kk < 8; kk += 2) {
            float4 p1[4];
            loadH(kbase + (kk + 1) * 32, p1);
            v8u w1 = loadW(kbase + (kk + 1) * 32);

            acc = wmma_bf16(packA(p0), w0, acc);

            if (kk + 2 < 8) {
                loadH(kbase + (kk + 2) * 32, p0);
                w0 = loadW(kbase + (kk + 2) * 32);
            }

            acc = wmma_bf16(packA(p1), w1, acc);
        }
    }
#pragma unroll
    for (int r = 0; r < 8; ++r) red[w][lane][r] = acc[r];
    __syncthreads();

    if (w == 0) {
#pragma unroll
        for (int r = 0; r < 8; ++r) {
            float s = red[0][lane][r] + red[1][lane][r]
                    + red[2][lane][r] + red[3][lane][r];
            int row = mt * 16 + r + 8 * hi;
            int col = nt * 16 + l15;
            float v = xp[(size_t)row * DH + col] + s;
            v = act ? tanhf(v) : fmaxf(v, 0.0f);
            xp[(size_t)row * DH + col] = v;
        }
    }
}

// ---------------------------------------------------------------------------
// Host launch
// ---------------------------------------------------------------------------
extern "C" void kernel_launch(void* const* d_in, const int* in_sizes, int n_in,
                              void* d_out, int out_size, void* d_ws, size_t ws_size,
                              hipStream_t stream) {
    (void)in_sizes; (void)n_in; (void)out_size; (void)ws_size;

    const float* x = (const float*)d_in[0];
    const float* Wih[3], * Whh[3], * bih[3], * bhh[3];
    for (int l = 0; l < 3; ++l) {
        Wih[l] = (const float*)d_in[1 + 4 * l];
        Whh[l] = (const float*)d_in[2 + 4 * l];
        bih[l] = (const float*)d_in[3 + 4 * l];
        bhh[l] = (const float*)d_in[4 + 4 * l];
    }

    // Workspace carve-up
    char* ws = (char*)d_ws;
    float* xp_buf = (float*)ws;                                    // 128 MB
    size_t off = (size_t)MTOT * DH * sizeof(float);
    unsigned short* xbf = (unsigned short*)(ws + off);             // 64 MB
    off += (size_t)MTOT * DH * sizeof(unsigned short);
    unsigned short* wih_bf[3], * whh_bf[3];
    for (int l = 0; l < 3; ++l) {
        wih_bf[l] = (unsigned short*)(ws + off); off += (size_t)DH * DH * 2;
        whh_bf[l] = (unsigned short*)(ws + off); off += (size_t)DH * DH * 2;
    }
    float* bsum[3];
    for (int l = 0; l < 3; ++l) {
        bsum[l] = (float*)(ws + off); off += DH * sizeof(float);
    }

    // Stage weights + biases (every call; deterministic)
    for (int l = 0; l < 3; ++l) {
        cvt_f32_bf16<<<1024, 256, 0, stream>>>(Wih[l], (unsigned*)wih_bf[l], DH * DH / 2);
        cvt_f32_bf16<<<1024, 256, 0, stream>>>(Whh[l], (unsigned*)whh_bf[l], DH * DH / 2);
        bias_sum<<<DH / 256, 256, 0, stream>>>(bih[l], bhh[l], bsum[l]);
    }

    for (int l = 0; l < 3; ++l) {
        const float* in = (l == 0) ? x : xp_buf;
        float* outb = (l == 2) ? (float*)d_out : xp_buf;

        // bf16 copy of layer input
        cvt_f32_bf16<<<8192, 256, 0, stream>>>(in, (unsigned*)xbf, MTOT * DH / 2);

        // xp = in * Wih^T + (bih + bhh)   -> 32768 waves, 8 per block
        proj_gemm<<<4096, 256, 0, stream>>>(xbf, wih_bf[l], bsum[l], outb);

        // sequential recurrence, in-place over xp
        const int act = (l == 0) ? 0 : 1;
        for (int t = 0; t < SEQ; ++t)
            rnn_step<<<256, 128, 0, stream>>>(whh_bf[l], outb, t, act);
    }
}
